// ZZYAttention_64639257804909
// MI455X (gfx1250) — matrix operand order repaired and checked by reading, not compiled, not run on hardware
//
#include <hip/hip_runtime.h>

typedef __attribute__((ext_vector_type(16))) _Float16 v16h;
typedef __attribute__((ext_vector_type(8)))  float    v8f;
typedef __attribute__((ext_vector_type(4)))  float    vf4;

namespace {
constexpr int kB = 2, kL = 2048, kS = 2048, kH = 16, kD = 64;
constexpr int kTileRows = 16;
constexpr int kTilesL   = kL / kTileRows;   // 128 row tiles per (b,h)
constexpr int kHD       = kH * kD;          // 1024 floats: s-stride inside values
constexpr int kScPitch  = 33;               // padded LDS pitch (bank-conflict free)
constexpr int kVvPitch  = 65;
}

__launch_bounds__(32)
__global__ void zzy_attn_fused(const float* __restrict__ scores,
                               const float* __restrict__ values,
                               float* __restrict__ out)
{
  __shared__ float sc_lds[kTileRows * kScPitch]; // 16 x 32 score chunk (+pad)
  __shared__ float vv_lds[32 * kVvPitch];        // 32 x 64 values chunk (+pad)

  const int lane = threadIdx.x;   // 0..31
  const int rL   = lane & 15;     // row for A/stats; N index for B/C
  const int hi   = lane >> 4;     // which half of the K striping this lane owns

  const int tile = blockIdx.x;
  const int lt = tile % kTilesL;
  const int bh = tile / kTilesL;
  const int h  = bh % kH;
  const int b  = bh / kH;
  const int l0 = lt * kTileRows;

  const float* sc_base = scores + ((size_t)(b * kH + h) * kL + l0) * kS;
  const float* v_base  = values + (size_t)b * kS * kHD + h * kD;

  // ---- Pass 1: softmax-1 stats (max & sum-exp) over the FULL S axis ----
  float m_r = 0.f, inv_s1 = 0.f;   // kept by the two lanes owning row rL
  for (int r = 0; r < kTileRows; ++r) {
    const float* row = sc_base + (size_t)r * kS;
    float m = -3.402823e38f, ss = 0.f;
    for (int i = lane; i < kS / 4; i += 32) {       // coalesced float4 stream
      vf4 x = *(const vf4*)(row + i * 4);
      float mx = fmaxf(fmaxf(x.x, x.y), fmaxf(x.z, x.w));
      float mn = fmaxf(m, mx);
      ss = ss * __expf(m - mn)
         + __expf(x.x - mn) + __expf(x.y - mn)
         + __expf(x.z - mn) + __expf(x.w - mn);
      m = mn;
    }
    #pragma unroll
    for (int off = 16; off > 0; off >>= 1) {        // wave32 butterfly
      float mo = __shfl_xor(m, off, 32);
      float so = __shfl_xor(ss, off, 32);
      float mn = fmaxf(m, mo);
      ss = ss * __expf(m - mn) + so * __expf(mo - mn);
      m = mn;
    }
    if (rL == r) { m_r = m; inv_s1 = 1.0f / ss; }
  }

  // ---- Pass 2: masked exp(w) fused with A@V via f16 WMMA, fp32 accum ----
  v8f c[4] = {};                 // 4 N-tiles of 16 -> D=64
  float denom = 0.f;             // per-row softmax-2 denominator (partial)
  const int kb = hi * 8;         // A-frag K base for this half-wave
  const int kh = hi * 16;        // unused: B frags use kb, same K order as A
  const int nchunk = (l0 + kTileRows + 31) >> 5;   // causal: only K <= l0+15
  for (int ch = 0; ch < nchunk; ++ch) {
    const int s0 = ch * 32;
    // stage score chunk [16 x 32] (coalesced 128B per row)
    for (int r = 0; r < kTileRows; ++r)
      sc_lds[r * kScPitch + lane] = sc_base[(size_t)r * kS + s0 + lane];
    // stage values chunk [32 x 64] (coalesced 256B per row)
    {
      const float* vp = v_base + (size_t)s0 * kHD + 2 * lane;
      #pragma unroll 4
      for (int k = 0; k < 32; ++k) {
        float2 g = *(const float2*)(vp + k * kHD);
        vv_lds[k * kVvPitch + 2 * lane]     = g.x;
        vv_lds[k * kVvPitch + 2 * lane + 1] = g.y;
      }
    }
    __syncthreads();

    // A fragment: 16-bit 16x32 layout (lane rL, K = kb+{0..7} and kb+16..23)
    v16h a;
    #pragma unroll
    for (int j = 0; j < 8; ++j) {
      const int koff = ((j >> 2) * 16) + kb + 2 * (j & 3);
      const float x0 = sc_lds[rL * kScPitch + koff];
      const float x1 = sc_lds[rL * kScPitch + koff + 1];
      const int sA = s0 + koff;
      // w = softmax-1 prob; e = exp(w) (w in [0,1] -> no max shift needed);
      // causal mask: zero contributes nothing to matmul nor denominator.
      const float e0 = (sA     <= l0 + rL) ? __expf(__expf(x0 - m_r) * inv_s1) : 0.f;
      const float e1 = (sA + 1 <= l0 + rL) ? __expf(__expf(x1 - m_r) * inv_s1) : 0.f;
      denom += e0 + e1;
      a[2 * j]     = (_Float16)e0;
      a[2 * j + 1] = (_Float16)e1;
    }

    // B fragments (32x16 per N-tile) from LDS, then WMMA accumulate
    #pragma unroll
    for (int t = 0; t < 4; ++t) {
      v16h bm;
      #pragma unroll
      for (int j = 0; j < 8; ++j) {
        const int k = ((j >> 2) * 16) + kb + 2 * (j & 3);
        bm[2 * j]     = (_Float16)vv_lds[k * kVvPitch + t * 16 + rL];
        bm[2 * j + 1] = (_Float16)vv_lds[(k + 1) * kVvPitch + t * 16 + rL];
      }
      c[t] = __builtin_amdgcn_wmma_f32_16x16x32_f16(
          /*neg_a=*/false, a, /*neg_b=*/false, bm,
          /*c_mod=*/(short)0, c[t], /*reuse_a=*/false, /*reuse_b=*/false);
    }
    __syncthreads();   // protect LDS before next chunk overwrites it
  }

  // combine the two half-lane partials of each row's denominator
  denom += __shfl_xor(denom, 16, 32);
  const float invd = 1.0f / denom;

  // ---- scale + store: C layout VGPR i -> row (i + 8*hi), col rL (+16t) ----
  #pragma unroll
  for (int i = 0; i < 8; ++i) {
    const int row = i + 8 * hi;
    const float sc = __shfl(invd, row, 32);   // lane 'row' holds that row's 1/denom
    float* po = out + ((size_t)(b * kL + l0 + row) * kH + h) * kD + rL;
    #pragma unroll
    for (int t = 0; t < 4; ++t)
      po[t * 16] = c[t][i] * sc;
  }
}

extern "C" void kernel_launch(void* const* d_in, const int* in_sizes, int n_in,
                              void* d_out, int out_size, void* d_ws, size_t ws_size,
                              hipStream_t stream) {
  (void)in_sizes; (void)n_in; (void)d_ws; (void)ws_size; (void)out_size;
  // setup_inputs order: queries(0), keys(1), values(2), scores(3); q/k unused.
  const float* values = (const float*)d_in[2];
  const float* scores = (const float*)d_in[3];
  float* out = (float*)d_out;
  const int grid = kB * kH * kTilesL;   // 2*16*128 = 4096 workgroups, 1 wave each
  hipLaunchKernelGGL(zzy_attn_fused, dim3(grid), dim3(32), 0, stream,
                     scores, values, out);
}
